// EcgGCNTCNModel_3994319585590
// MI455X (gfx1250) — matrix operand order repaired and checked by reading, not compiled, not run on hardware
//
#include <hip/hip_runtime.h>
#include <hip/hip_bf16.h>
#include <math.h>

// ---------------------------------------------------------------------------
// ECG GCN-TCN model for MI455X (gfx1250, wave32, WMMA).
// All heavy GEMMs use v_wmma_f32_16x16x32_bf16 (bf16 A/B, f32 accum).
// Convs are lowered to implicit GEMM (im2col with K = ci*2 + jj so the
// (cout,cin,2) conv weight is directly the [N][K] f32 operand; the GEMM
// converts weights f32->bf16 while staging into LDS).
// A-operand staging uses gfx1250 async global->LDS copies when available.
// ---------------------------------------------------------------------------

typedef __bf16 bf16_t;
typedef __attribute__((ext_vector_type(16))) __bf16 v16bf;
typedef __attribute__((ext_vector_type(8)))  float  v8f;
typedef __attribute__((ext_vector_type(4)))  int    v4i_t;
typedef __attribute__((address_space(1))) v4i_t gv4i_t;   // global v4i
typedef __attribute__((address_space(3))) v4i_t lv4i_t;   // LDS v4i

#if __has_builtin(__builtin_amdgcn_global_load_async_to_lds_b128) && \
    __has_builtin(__builtin_amdgcn_s_wait_asynccnt)
#define HAVE_ASYNC_LDS 1
#else
#define HAVE_ASYNC_LDS 0
#endif

#define B_SZ    64
#define LEADSN  12
#define SEQLEN  2048
#define STEPN   256
#define NSTEP   8
#define NCLSN   9
#define NBLOCKS 16
#define NBATCH  (NSTEP * B_SZ)      // 512 (outer-step, batch) samples
#define MTCN    (NBATCH * STEPN)    // 131072 rows for TCN GEMMs
#define EPSV    1e-5f

__device__ __forceinline__ float bn_scale(float g) { return g * rsqrtf(1.0f + EPSV); }

// ---------------------------------------------------------------------------
// Generic batched WMMA GEMM:  C[M,N] = act(A_bf16[M,K] @ B_f32 + bias)
//   bmode 0: B is f32 [N][K] (weight W, computes A @ W^T)
//   bmode 1: B is f32 [K][N]
// Block: 128 threads = 4 wave32; block tile 32(M) x 64(N); K chunk 32.
// Each wave owns one 16-wide N column pair of tiles: 2 WMMAs per K chunk,
// sharing one B fragment. Fragments are read from LDS as 2x ds_load_b128.
// A tile staging: async global->LDS b128 (ASYNCcnt) when rows are 16B-aligned.
// ---------------------------------------------------------------------------
__global__ void k_gemm(const bf16_t* __restrict__ A, const float* __restrict__ Bm,
                       const float* __restrict__ bias, float* __restrict__ C,
                       int M, int N, int K,
                       long aStride, long bStride, long cStride, int bBatchMod,
                       int bmode, int act)
{
    __shared__ __align__(16) bf16_t sA[32 * 32];   // [m][k]
    __shared__ __align__(16) bf16_t sB[64 * 32];   // [n][k]  (frag-contiguous)

    int batch = blockIdx.z;
    int m0 = blockIdx.x * 32;
    int n0 = blockIdx.y * 64;
    const bf16_t* Ab = A + (long)batch * aStride;
    int bIdx = bBatchMod ? (batch % bBatchMod) : batch;
    const float* Bb = Bm + (long)bIdx * bStride;
    float* Cb = C + (long)batch * cStride;

    int tid  = threadIdx.x;
    int wave = tid >> 5;
    int lane = tid & 31;

    // Fragment geometry per CDNA5 ISA VGPR layout tables (wave32).
    int mrow  = lane & 15;                 // A row within 16-row subtile
    int khalf = (lane & 16) ? 8 : 0;       // A K sub-half for lanes 16-31
    int kbase = (lane & 16) ? 8 : 0;       // B K sub-half for lanes 16-31
    int ncol  = (wave << 4) + (lane & 15); // N within block tile

    union Frag { v16bf v; uint4 q[2]; };

    v8f acc0 = {};
    v8f acc1 = {};
    for (int k0 = 0; k0 < K; k0 += 32) {
        bool fullK = (k0 + 32 <= K);
        bool fullA = fullK && (m0 + 32 <= M);
        // ---- stage A tile 32x32 (issued first so async copy overlaps B) ----
#if HAVE_ASYNC_LDS
        if (fullA && ((K & 7) == 0)) {
            // 128 threads x 16B = 2KB tile; rows 16B-aligned since K%8==0.
            int r = tid >> 2, c16 = (tid & 3) << 3;   // 8 bf16 per thread
            const bf16_t* gp = Ab + (long)(m0 + r) * K + k0 + c16;
            __builtin_amdgcn_global_load_async_to_lds_b128(
                (gv4i_t*)gp, (lv4i_t*)&sA[r * 32 + c16], 0, 0);
        } else
#endif
        if (fullA) {
#pragma unroll
            for (int j = 0; j < 2; ++j) {                 // 2x uint2 per thread
                int q = tid + j * 128;
                int r = q >> 3, cq = (q & 7) << 2;        // 4 bf16 per load
                *reinterpret_cast<uint2*>(&sA[r * 32 + cq]) =
                    *reinterpret_cast<const uint2*>(Ab + (long)(m0 + r) * K + k0 + cq);
            }
        } else {
            for (int idx = tid; idx < 32 * 32; idx += 128) {
                int r = idx >> 5, c = idx & 31;
                int gm = m0 + r, gk = k0 + c;
                sA[idx] = (gm < M && gk < K) ? Ab[(long)gm * K + gk] : (bf16_t)0.0f;
            }
        }
        // ---- stage B tile (k,n) -> LDS [n][k], f32 -> bf16 on the fly ----
        if (bmode == 0) {
            if (fullK && (n0 + 64 <= N)) {
#pragma unroll
                for (int j = 0; j < 4; ++j) {             // 4x float4 per thread
                    int q = tid + j * 128;
                    int n = q >> 3, ck = (q & 7) << 2;
                    float4 v = *reinterpret_cast<const float4*>(
                        Bb + (long)(n0 + n) * K + k0 + ck);
                    union { bf16_t h[4]; uint2 u; } t;
                    t.h[0] = (bf16_t)v.x; t.h[1] = (bf16_t)v.y;
                    t.h[2] = (bf16_t)v.z; t.h[3] = (bf16_t)v.w;
                    *reinterpret_cast<uint2*>(&sB[n * 32 + ck]) = t.u;
                }
            } else {
                for (int idx = tid; idx < 32 * 64; idx += 128) {
                    int kk = idx & 31, nn = idx >> 5;
                    int gk = k0 + kk, gn = n0 + nn;
                    float v = (gk < K && gn < N) ? Bb[(long)gn * K + gk] : 0.0f;
                    sB[nn * 32 + kk] = (bf16_t)v;
                }
            }
        } else {
            if (fullK && (n0 + 64 <= N)) {
#pragma unroll
                for (int j = 0; j < 4; ++j) {
                    int q = tid + j * 128;
                    int kk = q >> 4, cn = (q & 15) << 2;
                    float4 v = *reinterpret_cast<const float4*>(
                        Bb + (long)(k0 + kk) * N + n0 + cn);
                    sB[(cn + 0) * 32 + kk] = (bf16_t)v.x;
                    sB[(cn + 1) * 32 + kk] = (bf16_t)v.y;
                    sB[(cn + 2) * 32 + kk] = (bf16_t)v.z;
                    sB[(cn + 3) * 32 + kk] = (bf16_t)v.w;
                }
            } else {
                for (int idx = tid; idx < 32 * 64; idx += 128) {
                    int kk = idx >> 6, nn = idx & 63;
                    int gk = k0 + kk, gn = n0 + nn;
                    float v = (gk < K && gn < N) ? Bb[(long)gk * N + gn] : 0.0f;
                    sB[nn * 32 + kk] = (bf16_t)v;
                }
            }
        }
#if HAVE_ASYNC_LDS
        __builtin_amdgcn_s_wait_asynccnt(0);   // own async copies landed in LDS
#endif
        __syncthreads();

        if (k0 + 32 < K)  // hint next A chunk -> global_prefetch_b8
            __builtin_prefetch(Ab + (long)(m0 + (tid & 15)) * K + k0 + 32, 0, 1);

        // ---- fragments: 2x ds_load_b128 each (layouts are frag-contiguous) ----
        Frag a0, a1, bfv;
        {
            const uint4* pa0 = reinterpret_cast<const uint4*>(&sA[mrow * 32 + khalf]);
            a0.q[0] = pa0[0];          // k = khalf .. khalf+7
            a0.q[1] = pa0[2];          // k = 16+khalf .. 16+khalf+7
            const uint4* pa1 = reinterpret_cast<const uint4*>(&sA[(16 + mrow) * 32 + khalf]);
            a1.q[0] = pa1[0];
            a1.q[1] = pa1[2];
            const uint4* pb = reinterpret_cast<const uint4*>(&sB[ncol * 32 + kbase]);
            bfv.q[0] = pb[0];          // k = kbase .. kbase+7
            bfv.q[1] = pb[2];          // k = 16+kbase .. 16+kbase+7
        }
        acc0 = __builtin_amdgcn_wmma_f32_16x16x32_bf16(false, a0.v, false, bfv.v,
                                                       (short)0, acc0, false, false);
        acc1 = __builtin_amdgcn_wmma_f32_16x16x32_bf16(false, a1.v, false, bfv.v,
                                                       (short)0, acc1, false, false);
        __syncthreads();
    }

    // Store C (lane r: M = r + 8*(lane>=16), N = ncol) + bias + activation.
    int gn = n0 + ncol;
    if (gn < N) {
        float bv = bias ? bias[gn] : 0.0f;
#pragma unroll
        for (int r = 0; r < 8; ++r) {
            int gm = m0 + r + ((lane & 16) ? 8 : 0);
            if (gm < M) {
                float v = acc0[r] + bv;
                if (act == 1)      v = fmaxf(v, 0.0f);
                else if (act == 2) v = (v > 0.0f) ? v : 0.01f * v;
                Cb[(long)gm * N + gn] = v;
            }
            int gm1 = gm + 16;
            if (gm1 < M) {
                float v = acc1[r] + bv;
                if (act == 1)      v = fmaxf(v, 0.0f);
                else if (act == 2) v = (v > 0.0f) ? v : 0.01f * v;
                Cb[(long)gm1 * N + gn] = v;
            }
        }
    }
}

// ----------------------------- small kernels -------------------------------

__global__ void k_zero_f32(float* p, int n) {
    int i = blockIdx.x * 256 + threadIdx.x; if (i < n) p[i] = 0.0f;
}
__global__ void k_zero_b16(bf16_t* p, int n) {
    int i = blockIdx.x * 256 + threadIdx.x; if (i < n) p[i] = (bf16_t)0.0f;
}
__global__ void k_f2b(const float* __restrict__ in, bf16_t* __restrict__ out, int n) {
    int i = blockIdx.x * 256 + threadIdx.x; if (i < n) out[i] = (bf16_t)in[i];
}

// Build GRU input rows for all 8 outer steps x 12 lead-steps at once:
// row = (i*12 + t)*64 + b ; A[row][k] = x[b][t][i*256 + k]
__global__ void k_gipack(const float* __restrict__ x, bf16_t* __restrict__ out) {
    int idx = blockIdx.x * 256 + threadIdx.x;   // 6144*256
    int k = idx & 255;
    int row = idx >> 8;
    int b = row & 63;
    int it = row >> 6;
    int i = it / 12, t = it % 12;
    out[idx] = (bf16_t)x[((long)b * LEADSN + t) * SEQLEN + i * STEPN + k];
}

// GRU gate fusion + state update for all 512 (i,b) rows at lead-step t.
__global__ void k_grupoint(const float* __restrict__ gi, const float* __restrict__ gh,
                           float* __restrict__ hf, bf16_t* __restrict__ hb,
                           float* __restrict__ hseq, int t)
{
    int s = blockIdx.x;        // (i*64 + b)
    int n = threadIdx.x;       // 0..255
    int i = s >> 6, b = s & 63;
    long girow = (long)((i * 12 + t) * 64 + b) * 768;
    long ghrow = (long)s * 768;
    float r  = 1.0f / (1.0f + __expf(-(gi[girow + n]       + gh[ghrow + n])));
    float z  = 1.0f / (1.0f + __expf(-(gi[girow + 256 + n] + gh[ghrow + 256 + n])));
    float nn = tanhf(gi[girow + 512 + n] + r * gh[ghrow + 512 + n]);
    float h  = hf[s * 256 + n];
    h = (1.0f - z) * nn + z * h;
    hf[s * 256 + n] = h;
    hb[s * 256 + n] = (bf16_t)h;
    hseq[((long)s * 12 + t) * 256 + n] = h;
}

// Graph learning per (outer step, batch): kqv einsums (12x12), column softmax,
// relu(score@value), symmetrize, D^-1/2 A D^-1/2.
__global__ void k_graph(const float* __restrict__ hseq, const float* __restrict__ kw,
                        const float* __restrict__ qw, const float* __restrict__ vw,
                        float* __restrict__ adj)
{
    __shared__ float sh[12 * 256];
    __shared__ float kqv[3][12][12];
    __shared__ float sc[12][12];
    __shared__ float sa[12][12];
    __shared__ float dd[12];
    int s = blockIdx.x; int b = s & 63;
    int tid = threadIdx.x;  // 128
    for (int idx = tid; idx < 12 * 256; idx += 128)
        sh[idx] = hseq[(long)s * 12 * 256 + idx];
    __syncthreads();
    for (int idx = tid; idx < 432; idx += 128) {
        int which = idx / 144, rem = idx % 144;
        int l = rem / 12, n = rem % 12;
        const float* w = (which == 0 ? kw : (which == 1 ? qw : vw)) + (long)b * 256 * 12;
        float acc = 0.0f;
        for (int sd = 0; sd < 256; ++sd) acc += sh[l * 256 + sd] * w[sd * 12 + n];
        kqv[which][l][n] = acc;
    }
    __syncthreads();
    for (int idx = tid; idx < 144; idx += 128) {
        int l = idx / 12, m = idx % 12;
        float acc = 0.0f;
        for (int n = 0; n < 12; ++n) acc += kqv[1][l][n] * kqv[0][m][n]; // q.k
        sc[l][m] = acc;
    }
    __syncthreads();
    if (tid < 12) {             // softmax over axis=1 (over l, per column m)
        int m = tid; float mx = -1e30f;
        for (int l = 0; l < 12; ++l) mx = fmaxf(mx, sc[l][m]);
        float sum = 0.0f;
        for (int l = 0; l < 12; ++l) { float e = __expf(sc[l][m] - mx); sc[l][m] = e; sum += e; }
        float inv = 1.0f / sum;
        for (int l = 0; l < 12; ++l) sc[l][m] *= inv;
    }
    __syncthreads();
    for (int idx = tid; idx < 144; idx += 128) {
        int l = idx / 12, n = idx % 12;
        float acc = 0.0f;
        for (int m = 0; m < 12; ++m) acc += sc[l][m] * kqv[2][m][n];
        sa[l][n] = fmaxf(acc, 0.0f);
    }
    __syncthreads();
    for (int idx = tid; idx < 144; idx += 128) {
        int l = idx / 12, n = idx % 12;
        sc[l][n] = 0.5f * (sa[l][n] + sa[n][l]);
    }
    __syncthreads();
    if (tid < 12) {
        int n = tid; float rs = 0.0f;
        for (int l = 0; l < 12; ++l) rs += sc[l][n];
        dd[n] = (rs > 0.0f) ? rsqrtf(rs) : 0.0f;
    }
    __syncthreads();
    for (int idx = tid; idx < 144; idx += 128) {
        int l = idx / 12, n = idx % 12;
        adj[(long)s * 144 + idx] = dd[l] * sc[l][n] * dd[n];
    }
}

// x2 = adj @ xx and transpose to (sample, time, channel) bf16 for implicit GEMM.
__global__ void k_x2t(const float* __restrict__ adj, const float* __restrict__ x,
                      bf16_t* __restrict__ out)
{
    __shared__ float sadj[144];
    int s = blockIdx.x; int t = threadIdx.x;   // 512 blocks x 256
    int i = s >> 6, b = s & 63;
    if (t < 144) sadj[t] = adj[(long)s * 144 + t];
    __syncthreads();
    float xv[12];
    for (int m = 0; m < 12; ++m)
        xv[m] = x[((long)b * 12 + m) * SEQLEN + i * STEPN + t];
    for (int c = 0; c < 12; ++c) {
        float acc = 0.0f;
        for (int m = 0; m < 12; ++m) acc += sadj[c * 12 + m] * xv[m];
        out[((long)s * STEPN + t) * 12 + c] = (bf16_t)acc;
    }
}

// im2col for causal dilated conv k=2: col[m][ci*2+jj] = in[t-(1-jj)*dil][ci]
__global__ void k_im2col(const bf16_t* __restrict__ in, bf16_t* __restrict__ col,
                         int cin, int dil)
{
    int m = blockIdx.x * 256 + threadIdx.x;   // < MTCN
    int s = m >> 8, t = m & 255;
    const bf16_t* base = in + (long)s * STEPN * cin;
    bf16_t* o = col + (long)m * 2 * cin;
    for (int ci = 0; ci < cin; ++ci) {
        o[ci * 2]     = (t >= dil) ? base[(t - dil) * cin + ci] : (bf16_t)0.0f;
        o[ci * 2 + 1] = base[t * cin + ci];
    }
}

// out = relu(conv2_out + residual), residual either f32 (1x1 conv) or bf16 (identity).
__global__ void k_addres(const float* __restrict__ out2, const float* __restrict__ resf,
                         const bf16_t* __restrict__ resb, bf16_t* __restrict__ out,
                         int n, int useBf)
{
    int i = blockIdx.x * 256 + threadIdx.x;
    if (i >= n) return;
    float r = useBf ? (float)resb[i] : resf[i];
    out[i] = (bf16_t)fmaxf(out2[i] + r, 0.0f);
}

// TCN output (512,256,12) -> (512,12,256) bf16 rows for the batched g GEMM.
__global__ void k_tpack(const bf16_t* __restrict__ in, bf16_t* __restrict__ out) {
    int idx = blockIdx.x * 256 + threadIdx.x;  // 512*12*256
    int tt = idx & 255; int r = idx >> 8; int l = r % 12; int s = r / 12;
    out[idx] = in[((long)s * STEPN + tt) * 12 + l];
}

// BN (eval form) fused with f32->bf16 pack; channel = (row % 12).
__global__ void k_bnpack(const float* __restrict__ in, const float* __restrict__ g,
                         const float* __restrict__ bb, bf16_t* __restrict__ out, int n)
{
    int i = blockIdx.x * 256 + threadIdx.x;
    if (i >= n) return;
    int l = (i >> 8) % 12;
    out[i] = (bf16_t)(in[i] * bn_scale(g[l]) + bb[l]);
}

// ResNet conv7-same + BN (+residual) + relu; 12 channels -> scalar VALU path.
__global__ void k_resnet(const float* __restrict__ xin, const float* __restrict__ resid,
                         float* __restrict__ xout, const float* __restrict__ w,
                         const float* __restrict__ gamma, const float* __restrict__ beta,
                         int mode)
{
    int g = blockIdx.x * 256 + threadIdx.x;   // 64*2048
    int b = g >> 11, t = g & 2047;
    const float* xb = xin + (long)b * 12 * SEQLEN;
    for (int co = 0; co < 12; ++co) {
        float acc = 0.0f;
        const float* wc = w + co * 84;
        for (int ci = 0; ci < 12; ++ci) {
            const float* xr = xb + ci * SEQLEN;
#pragma unroll
            for (int k = 0; k < 7; ++k) {
                int tt = t + k - 3;
                if (tt >= 0 && tt < SEQLEN) acc += wc[ci * 7 + k] * xr[tt];
            }
        }
        float v = acc * bn_scale(gamma[co]) + beta[co];
        if (mode == 2) v += resid[((long)b * 12 + co) * SEQLEN + t];
        xout[((long)b * 12 + co) * SEQLEN + t] = fmaxf(v, 0.0f);
    }
}

// res = concat(block outputs) + resnet_out, packed straight to the FC1 A matrix.
__global__ void k_combine(const float* __restrict__ OUTf, const float* __restrict__ rn,
                          bf16_t* __restrict__ resb)
{
    int idx = blockIdx.x * 256 + threadIdx.x;  // 64*24576
    int b = idx / 24576; int rem = idx % 24576;
    int l = rem >> 11;   int p = rem & 2047;
    int i = p >> 8;      int tt = p & 255;
    float v = OUTf[(((long)(i * 64 + b) * 12) + l) * 256 + tt] + rn[idx];
    resb[idx] = (bf16_t)v;
}

__global__ void k_fc3(const float* __restrict__ h2, const float* __restrict__ w3,
                      const float* __restrict__ b3, float* __restrict__ out)
{
    int idx = blockIdx.x * 64 + threadIdx.x;
    if (idx >= 64 * NCLSN) return;
    int b = idx / NCLSN, c = idx % NCLSN;
    float acc = b3[c];
    const float* h = h2 + (long)b * 2048;
    const float* w = w3 + (long)c * 2048;
    for (int k = 0; k < 2048; ++k) acc += h[k] * w[k];
    out[idx] = acc;
}

// ------------------------------- host side ---------------------------------

static inline int cdiv(int a, int b) { return (a + b - 1) / b; }

static void gemm(hipStream_t st, const bf16_t* A, const float* B, const float* bias,
                 float* C, int M, int N, int K, int batches,
                 long aS, long bS, long cS, int bMod, int bmode, int act)
{
    dim3 g(cdiv(M, 32), cdiv(N, 64), batches);
    k_gemm<<<g, 128, 0, st>>>(A, B, bias, C, M, N, K, aS, bS, cS, bMod, bmode, act);
}

extern "C" void kernel_launch(void* const* d_in, const int* in_sizes, int n_in,
                              void* d_out, int out_size, void* d_ws, size_t ws_size,
                              hipStream_t stream)
{
    (void)in_sizes; (void)out_size; (void)ws_size;
    // Input order (jax pytree flatten; None leaves of tcn_sb dropped):
    // 0:x 1:w_ih 2:w_hh 3:b_ih 4:b_hh 5:kw 6:qw 7:vw 8:stock_weight
    // 9..14:  L0 w1,b1,w2,b2,dw,db   15..18: L1 w1,b1,w2,b2
    // 19..22: L2 w1,b1,w2,b2         23..28: L3 w1,b1,w2,b2,dw,db
    // 29..34: resnet w1s,w2s,g1s,b1s,g2s,b2s
    // 35..40: block lw1,lb1,bg,bb,lw2,lb2
    // 41..46: fc w1,b1,w2,b2,w3,b3
    const float* IN[47];
    for (int i = 0; i < 47 && i < n_in; ++i) IN[i] = (const float*)d_in[i];

    char* base = (char*)d_ws; size_t off = 0;
    auto alloc = [&](size_t bytes) -> void* {
        void* p = base + off; off += (bytes + 255) & ~(size_t)255; return p;
    };
    float*  gi    = (float*) alloc(6144UL * 768 * 4);
    bf16_t* Aall  = (bf16_t*)alloc(6144UL * 256 * 2);
    float*  hf    = (float*) alloc(512UL * 256 * 4);
    bf16_t* hb    = (bf16_t*)alloc(512UL * 256 * 2);
    float*  gh    = (float*) alloc(512UL * 768 * 4);
    float*  hseq  = (float*) alloc(512UL * 12 * 256 * 4);
    float*  adjb  = (float*) alloc(512UL * 144 * 4);
    bf16_t* x2t   = (bf16_t*)alloc(512UL * 256 * 12 * 2);
    bf16_t* colb  = (bf16_t*)alloc((size_t)MTCN * 256 * 2);
    float*  out1f = (float*) alloc((size_t)MTCN * 128 * 4);
    bf16_t* out1b = (bf16_t*)alloc((size_t)MTCN * 128 * 2);
    float*  out2f = (float*) alloc((size_t)MTCN * 128 * 4);
    float*  resf  = (float*) alloc((size_t)MTCN * 64 * 4);
    bf16_t* curA  = (bf16_t*)alloc((size_t)MTCN * 128 * 2);
    bf16_t* curB  = (bf16_t*)alloc((size_t)MTCN * 128 * 2);
    bf16_t* At    = (bf16_t*)alloc(512UL * 12 * 256 * 2);
    float*  G     = (float*) alloc(6144UL * 256 * 4);
    bf16_t* Gb    = (bf16_t*)alloc(6144UL * 256 * 2);
    float*  Hf    = (float*) alloc(6144UL * 256 * 4);
    bf16_t* Hb    = (bf16_t*)alloc(6144UL * 256 * 2);
    float*  OUTf  = (float*) alloc(6144UL * 256 * 4);
    float*  rb0   = (float*) alloc(64UL * 12 * 2048 * 4);
    float*  rb1   = (float*) alloc(64UL * 12 * 2048 * 4);
    float*  yb    = (float*) alloc(64UL * 12 * 2048 * 4);
    bf16_t* resb  = (bf16_t*)alloc(64UL * 24576 * 2);
    float*  h1f   = (float*) alloc(64UL * 2048 * 4);
    bf16_t* h1b   = (bf16_t*)alloc(64UL * 2048 * 2);
    float*  h2f   = (float*) alloc(64UL * 2048 * 4);

    const float* x = IN[0];

    // ---- ResNet branch (16 blocks, scalar conv7+bn+relu, ping-pong) ----
    {
        const float *w1s = IN[29], *w2s = IN[30], *g1s = IN[31],
                    *b1s = IN[32], *g2s = IN[33], *b2s = IN[34];
        const float* cur = x; float* rb[2] = {rb0, rb1}; int pp = 0;
        for (int blk = 0; blk < NBLOCKS; ++blk) {
            k_resnet<<<512, 256, 0, stream>>>(cur, nullptr, yb,
                w1s + blk * 1008, g1s + blk * 12, b1s + blk * 12, 1);
            k_resnet<<<512, 256, 0, stream>>>(yb, cur, rb[pp],
                w2s + blk * 1008, g2s + blk * 12, b2s + blk * 12, 2);
            cur = rb[pp]; pp ^= 1;
        }
        // resnet output lives at `cur`
        // ---- GRU: gi for all 96 steps in one WMMA GEMM, then 12x recurrence ----
        k_gipack<<<6144, 256, 0, stream>>>(x, Aall);
        gemm(stream, Aall, IN[1], IN[3], gi, 6144, 768, 256, 1, 0, 0, 0, 0, 0, 0);
        k_zero_f32<<<512, 256, 0, stream>>>(hf, 512 * 256);
        k_zero_b16<<<512, 256, 0, stream>>>(hb, 512 * 256);
        for (int t = 0; t < 12; ++t) {
            gemm(stream, hb, IN[2], IN[4], gh, 512, 768, 256, 1, 0, 0, 0, 0, 0, 0);
            k_grupoint<<<512, 256, 0, stream>>>(gi, gh, hf, hb, hseq, t);
        }

        // ---- Graph learning + x2 = adj @ xx (transposed to (s,t,c)) ----
        k_graph<<<512, 128, 0, stream>>>(hseq, IN[5], IN[6], IN[7], adjb);
        k_x2t<<<512, 256, 0, stream>>>(adjb, x, x2t);

        // ---- TCN: 4 levels of causal dilated conv k=2 as implicit WMMA GEMM ----
        struct Lv { const float *w1, *b1, *w2, *b2, *dw, *db; int cin, cout, dil; };
        Lv L[4] = {
            {IN[9],  IN[10], IN[11], IN[12], IN[13], IN[14],  12,  64, 1},
            {IN[15], IN[16], IN[17], IN[18], nullptr, nullptr, 64, 128, 2},
            {IN[19], IN[20], IN[21], IN[22], nullptr, nullptr,128, 128, 4},
            {IN[23], IN[24], IN[25], IN[26], IN[27], IN[28], 128,  12, 8},
        };
        const bf16_t* tin = x2t;
        bf16_t* touts[4] = {curA, curB, curA, curB};
        for (int l = 0; l < 4; ++l) {
            int cin = L[l].cin, cout = L[l].cout, dil = L[l].dil;
            int n1 = MTCN * cout;
            k_im2col<<<MTCN / 256, 256, 0, stream>>>(tin, colb, cin, dil);
            gemm(stream, colb, L[l].w1, L[l].b1, out1f, MTCN, cout, 2 * cin,
                 1, 0, 0, 0, 0, 0, 1);                       // + relu
            k_f2b<<<cdiv(n1, 256), 256, 0, stream>>>(out1f, out1b, n1);
            k_im2col<<<MTCN / 256, 256, 0, stream>>>(out1b, colb, cout, dil);
            gemm(stream, colb, L[l].w2, L[l].b2, out2f, MTCN, cout, 2 * cout,
                 1, 0, 0, 0, 0, 0, 0);
            if (L[l].dw) {                                   // 1x1-conv residual
                gemm(stream, tin, L[l].dw, L[l].db, resf, MTCN, cout, cin,
                     1, 0, 0, 0, 0, 0, 0);
                k_addres<<<cdiv(n1, 256), 256, 0, stream>>>(out2f, resf, nullptr,
                                                            touts[l], n1, 0);
            } else {                                         // identity residual
                k_addres<<<cdiv(n1, 256), 256, 0, stream>>>(out2f, nullptr, tin,
                                                            touts[l], n1, 1);
            }
            tin = touts[l];
        }

        // ---- g = t @ stock_weight (batched over 512, B operand indexed by b) ----
        k_tpack<<<6144, 256, 0, stream>>>(tin, At);
        gemm(stream, At, IN[8], nullptr, G, 12, 256, 256, NBATCH,
             12 * 256, 256L * 256, 12 * 256, 64, 1, 0);

        // ---- block_out: leaky(g@lw1+lb1) -> bn -> @lw2+lb2 ----
        k_f2b<<<cdiv(6144 * 256, 256), 256, 0, stream>>>(G, Gb, 6144 * 256);
        gemm(stream, Gb, IN[35], IN[36], Hf, 6144, 256, 256, 1, 0, 0, 0, 0, 0, 2);
        k_bnpack<<<cdiv(6144 * 256, 256), 256, 0, stream>>>(Hf, IN[37], IN[38],
                                                            Hb, 6144 * 256);
        gemm(stream, Hb, IN[39], IN[40], OUTf, 6144, 256, 256, 1, 0, 0, 0, 0, 0, 0);

        // ---- combine with resnet branch, then FC head ----
        k_combine<<<6144, 256, 0, stream>>>(OUTf, cur, resb);
    }
    gemm(stream, resb, IN[41], IN[42], h1f, 64, 2048, 24576, 1, 0, 0, 0, 0, 0, 2);
    k_f2b<<<cdiv(64 * 2048, 256), 256, 0, stream>>>(h1f, h1b, 64 * 2048);
    gemm(stream, h1b, IN[43], IN[44], h2f, 64, 2048, 2048, 1, 0, 0, 0, 0, 0, 2);
    k_fc3<<<cdiv(64 * NCLSN, 64), 64, 0, stream>>>(h2f, IN[45], IN[46], (float*)d_out);
}
